// InterFrameAttention_68332929679393
// MI455X (gfx1250) — matrix operand order repaired and checked by reading, not compiled, not run on hardware
//
#include <hip/hip_runtime.h>

// ---------------- types / constants ----------------
typedef unsigned short bf16_t; // raw bf16 bits
typedef __attribute__((ext_vector_type(8)))  __bf16 v8bf;
typedef __attribute__((ext_vector_type(16))) __bf16 v16bf;
typedef __attribute__((ext_vector_type(8)))  float  v8f;

#define NTOK   2304
#define CDIM   256
#define NH     8
#define HD     32
#define HIDDEN 1024
#define MDIM   128
#define MHD    16
#define BB     4
#define MROWS  (BB*NTOK)
#define HH     48
#define SCALE_Q 0.17677669529663687f /* 32^-0.5 */

#define WMMA_BF16(a,b,c) \
  __builtin_amdgcn_wmma_f32_16x16x32_bf16(false,(a),false,(b),(short)0,(c),false,false)

static __device__ __forceinline__ bf16_t f2bf(float f) {
  unsigned u = __float_as_uint(f);
  u += 0x7FFFu + ((u >> 16) & 1u);           // round-to-nearest-even
  return (bf16_t)(u >> 16);
}
static __device__ __forceinline__ float bf2f(bf16_t h) {
  return __uint_as_float(((unsigned)h) << 16);
}

// A-fragment (16x32 bf16, MxK): lane = M, two 16B chunks:
//   elems 0..7  = K [16*half, 16*half+8)
//   elems 8..15 = K [16*half+8, 16*half+16)
static __device__ __forceinline__ v16bf load_afrag(const bf16_t* row, int half) {
  v8bf a0 = *(const v8bf*)(row + 16*half);
  v8bf a1 = *(const v8bf*)(row + 16*half + 8);
  return __builtin_shufflevector(a0, a1, 0,1,2,3,4,5,6,7,8,9,10,11,12,13,14,15);
}

// ---------------- weight transpose to bf16 (N x K row-major = "Bt") ----------------
__global__ __launch_bounds__(256) void k_transpose(const float* __restrict__ W,
                                                   bf16_t* __restrict__ Wt,
                                                   int K, int N) {
  int idx = blockIdx.x * 256 + threadIdx.x;
  if (idx >= K * N) return;
  int n = idx / K, k = idx % K;
  Wt[idx] = f2bf(W[(size_t)k * N + n]);
}

// ---------------- LayerNorm of NCHW tokens -> X (f32 + bf16), (B,N,C) ----------------
__global__ __launch_bounds__(256) void k_norm_tokens(const float* __restrict__ feat0,
                                                     const float* __restrict__ feat1,
                                                     const float* __restrict__ g,
                                                     const float* __restrict__ bta,
                                                     float* __restrict__ Xf,
                                                     bf16_t* __restrict__ Xb) {
  __shared__ float red[256];
  int row = blockIdx.x;              // 0..MROWS-1
  int b = row / NTOK, n = row % NTOK;
  int c = threadIdx.x;
  const float* src = (b < 2) ? (feat0 + ((size_t)b*CDIM + c)*NTOK + n)
                             : (feat1 + ((size_t)(b-2)*CDIM + c)*NTOK + n);
  float x = *src;
  red[c] = x; __syncthreads();
  for (int s = 128; s > 0; s >>= 1) { if (c < s) red[c] += red[c+s]; __syncthreads(); }
  float mean = red[0] * (1.0f/CDIM);
  __syncthreads();
  float d = x - mean;
  red[c] = d * d; __syncthreads();
  for (int s = 128; s > 0; s >>= 1) { if (c < s) red[c] += red[c+s]; __syncthreads(); }
  float var = red[0] * (1.0f/CDIM);
  float y = d * rsqrtf(var + 1e-5f) * g[c] + bta[c];
  Xf[(size_t)row*CDIM + c] = y;
  Xb[(size_t)row*CDIM + c] = f2bf(y);
}

// ---------------- LayerNorm of row-major f32 -> bf16 ----------------
__global__ __launch_bounds__(256) void k_ln2(const float* __restrict__ in,
                                             const float* __restrict__ g,
                                             const float* __restrict__ bta,
                                             bf16_t* __restrict__ outb) {
  __shared__ float red[256];
  int row = blockIdx.x;
  int c = threadIdx.x;
  float x = in[(size_t)row*CDIM + c];
  red[c] = x; __syncthreads();
  for (int s = 128; s > 0; s >>= 1) { if (c < s) red[c] += red[c+s]; __syncthreads(); }
  float mean = red[0] * (1.0f/CDIM);
  __syncthreads();
  float d = x - mean;
  red[c] = d * d; __syncthreads();
  for (int s = 128; s > 0; s >>= 1) { if (c < s) red[c] += red[c+s]; __syncthreads(); }
  float var = red[0] * (1.0f/CDIM);
  outb[(size_t)row*CDIM + c] = f2bf(d * rsqrtf(var + 1e-5f) * g[c] + bta[c]);
}

// ---------------- coordinate embedding ce (N,128) f32 ----------------
__global__ __launch_bounds__(128) void k_ce(const float* __restrict__ corw,
                                            const float* __restrict__ corb,
                                            float* __restrict__ ce) {
  int n = blockIdx.x, j = threadIdx.x;
  int w = n % HH, h = n / HH;
  float x = -1.0f + 2.0f * (float)w / 47.0f;
  float y = -1.0f + 2.0f * (float)h / 47.0f;
  ce[(size_t)n*MDIM + j] = x * corw[j] + y * corw[MDIM + j] + corb[j];
}

// -------- build Vt (B,NH,48,N) bf16: rows 0..31 = V (frame-swapped), 32..47 = ce heads --------
__global__ __launch_bounds__(256) void k_build_vt(const bf16_t* __restrict__ KVb,
                                                  const float* __restrict__ ce,
                                                  bf16_t* __restrict__ Vt) {
  int d = blockIdx.x % 48;
  int h = (blockIdx.x / 48) % NH;
  int b = blockIdx.x / (48 * NH);
  int bsrc = b ^ 2;
  bf16_t* dst = Vt + (((size_t)(b*NH + h))*48 + d) * NTOK;
  for (int m = threadIdx.x; m < NTOK; m += 256) {
    bf16_t v;
    if (d < 32) v = KVb[((size_t)(bsrc*NTOK + m))*512 + 256 + h*HD + d];
    else        v = f2bf(ce[(size_t)m*MDIM + h*MHD + (d-32)]);
    dst[m] = v;
  }
}

// ---------------- blocked WMMA GEMM: C = A(MxK) * Bt(NxK)^T + bias (+resid) ----------------
// wave tile = 32 rows x 64 cols (2 A-frags x 4 B-frags -> 8 WMMA per k-step);
// block = 8 waves arranged 2x4 -> 64x256 block tile.
__global__ __launch_bounds__(256) void k_gemm(const bf16_t* __restrict__ A,
                                              const bf16_t* __restrict__ Bt,
                                              const float* __restrict__ bias,
                                              const float* __restrict__ resid,
                                              float* __restrict__ Cf,
                                              bf16_t* __restrict__ Cb,
                                              float* __restrict__ Cn,
                                              int M, int Nn, int K, int tokens) {
  int wave = threadIdx.x >> 5;
  int lane = threadIdx.x & 31;
  int ln = lane & 15, half = lane >> 4;
  int tm = blockIdx.y * 2 + (wave >> 2);   // 32-row tile index
  int tn = blockIdx.x * 4 + (wave & 3);    // 64-col tile index
  if (tn * 64 >= Nn) return;               // uniform per wave (small-N GEMMs)
  const bf16_t* arow0 = A  + (size_t)(tm*32 + ln) * K;
  const bf16_t* arow1 = arow0 + (size_t)16 * K;
  const bf16_t* brow  = Bt + (size_t)(tn*64 + ln) * K;
  v8f acc[2][4] = {};
  for (int k = 0; k < K; k += 32) {
    __builtin_prefetch(arow0 + k + 64, 0, 0);
    __builtin_prefetch(arow1 + k + 64, 0, 0);
    v16bf a0 = load_afrag(arow0 + k, half);
    v16bf a1 = load_afrag(arow1 + k, half);
    v16bf b0 = *(const v16bf*)(brow + k + 16*half);
    v16bf b1 = *(const v16bf*)(brow + (size_t)16*K + k + 16*half);
    v16bf b2 = *(const v16bf*)(brow + (size_t)32*K + k + 16*half);
    v16bf b3 = *(const v16bf*)(brow + (size_t)48*K + k + 16*half);
    acc[0][0] = WMMA_BF16(a0, b0, acc[0][0]);
    acc[0][1] = WMMA_BF16(a0, b1, acc[0][1]);
    acc[0][2] = WMMA_BF16(a0, b2, acc[0][2]);
    acc[0][3] = WMMA_BF16(a0, b3, acc[0][3]);
    acc[1][0] = WMMA_BF16(a1, b0, acc[1][0]);
    acc[1][1] = WMMA_BF16(a1, b1, acc[1][1]);
    acc[1][2] = WMMA_BF16(a1, b2, acc[1][2]);
    acc[1][3] = WMMA_BF16(a1, b3, acc[1][3]);
  }
  #pragma unroll
  for (int i = 0; i < 2; ++i) {
    #pragma unroll
    for (int j = 0; j < 4; ++j) {
      int col = tn*64 + j*16 + ln;
      float bv = bias[col];
      #pragma unroll
      for (int r = 0; r < 8; ++r) {
        int row = tm*32 + i*16 + r + 8*half;
        float v = acc[i][j][r] + bv;
        if (resid) v += resid[(size_t)row*Nn + col];
        if (Cf) Cf[(size_t)row*Nn + col] = v;
        if (Cb) Cb[(size_t)row*Nn + col] = f2bf(v);
        if (Cn) { int bb = row / tokens, n = row % tokens;
                  Cn[((size_t)bb*Nn + col)*tokens + n] = v; }
      }
    }
  }
}

// ---------------- fused flash attention + motion gather ----------------
// per wave: 16 queries, one head; V is 48-wide (32 value dims + 16 ce dims)
__global__ __launch_bounds__(256) void k_attn(const bf16_t* __restrict__ Qb,
                                              const bf16_t* __restrict__ KVb,
                                              const bf16_t* __restrict__ Vt,
                                              const float* __restrict__ ce,
                                              bf16_t* __restrict__ attnx,
                                              bf16_t* __restrict__ crevmce) {
  __shared__ bf16_t pst[8][16*32];     // per-wave P staging (16q x 32m)
  int wave = threadIdx.x >> 5;
  int lane = threadIdx.x & 31;
  int ln = lane & 15, half = lane >> 4;
  int b = blockIdx.z, h = blockIdx.y;
  int q0 = blockIdx.x * 128 + wave * 16;
  int bsrc = b ^ 2;                    // feat_t batch remap

  const bf16_t* qrow = Qb + ((size_t)(b*NTOK + q0 + ln))*CDIM + h*HD;
  v16bf qf = load_afrag(qrow, half);

  float mrow[8], lrow[8];
  #pragma unroll
  for (int r = 0; r < 8; ++r) { mrow[r] = -1e30f; lrow[r] = 0.0f; }
  v8f o0 = {}, o1 = {}, o2 = {};

  const bf16_t* kbase = KVb + ((size_t)bsrc*NTOK)*512 + h*HD + 16*half;
  const bf16_t* vbase = Vt + ((size_t)(b*NH + h))*48*NTOK + (size_t)ln*NTOK + 16*half;

  for (int m0 = 0; m0 < NTOK; m0 += 32) {
    v16bf k0 = *(const v16bf*)(kbase + (size_t)(m0 + ln)*512);
    v16bf k1 = *(const v16bf*)(kbase + (size_t)(m0 + 16 + ln)*512);
    v8f s0 = {}, s1 = {};
    s0 = WMMA_BF16(qf, k0, s0);
    s1 = WMMA_BF16(qf, k1, s1);
    #pragma unroll
    for (int r = 0; r < 8; ++r) {
      float e0 = s0[r]*SCALE_Q, e1 = s1[r]*SCALE_Q;
      float t = fmaxf(e0, e1);
      t = fmaxf(t, __shfl_xor(t, 1, 32));
      t = fmaxf(t, __shfl_xor(t, 2, 32));
      t = fmaxf(t, __shfl_xor(t, 4, 32));
      t = fmaxf(t, __shfl_xor(t, 8, 32));       // row max within half-wave
      float mn = fmaxf(mrow[r], t);
      float alpha = __expf(mrow[r] - mn);
      mrow[r] = mn;
      e0 = __expf(e0 - mn); e1 = __expf(e1 - mn);
      float rs = e0 + e1;
      rs += __shfl_xor(rs, 1, 32);
      rs += __shfl_xor(rs, 2, 32);
      rs += __shfl_xor(rs, 4, 32);
      rs += __shfl_xor(rs, 8, 32);
      lrow[r] = lrow[r]*alpha + rs;
      o0[r] *= alpha; o1[r] *= alpha; o2[r] *= alpha;
      int row = r + 8*half;                     // C-layout row
      pst[wave][row*32 + ln]      = f2bf(e0);
      pst[wave][row*32 + 16 + ln] = f2bf(e1);
    }
    asm volatile("s_wait_dscnt 0" ::: "memory"); // CDNA5 split DS counter
    v16bf pf = load_afrag(&pst[wave][ln*32], half);
    v16bf v0 = *(const v16bf*)(vbase + m0);
    v16bf v1 = *(const v16bf*)(vbase + (size_t)16*NTOK + m0);
    v16bf v2 = *(const v16bf*)(vbase + (size_t)32*NTOK + m0);
    o0 = WMMA_BF16(pf, v0, o0);
    o1 = WMMA_BF16(pf, v1, o1);
    o2 = WMMA_BF16(pf, v2, o2);
  }
  #pragma unroll
  for (int r = 0; r < 8; ++r) {
    float invl = 1.0f / lrow[r];
    int rq = q0 + r + 8*half;
    size_t tok = (size_t)b*NTOK + rq;
    attnx[tok*CDIM + h*HD + ln]      = f2bf(o0[r]*invl);
    attnx[tok*CDIM + h*HD + 16 + ln] = f2bf(o1[r]*invl);
    float cev = ce[(size_t)rq*MDIM + h*MHD + ln];
    crevmce[tok*MDIM + h*MHD + ln]   = f2bf(o2[r]*invl - cev);
  }
}

// ---------------- depthwise 3x3 (SAME) + exact GELU: bf16 in -> bf16 out ----------------
__global__ __launch_bounds__(256) void k_dwconv_gelu(const bf16_t* __restrict__ y,
                                                     const float* __restrict__ dww,
                                                     const float* __restrict__ dwb,
                                                     bf16_t* __restrict__ g) {
  int bn = blockIdx.x;                 // 0..MROWS-1
  int b = bn / NTOK, n = bn % NTOK;
  int hh = n / HH, w = n % HH;
  #pragma unroll
  for (int i = 0; i < 4; ++i) {
    int c = threadIdx.x + 256*i;
    float acc = dwb[c];
    #pragma unroll
    for (int kh = -1; kh <= 1; ++kh) {
      int h2 = hh + kh; if (h2 < 0 || h2 >= HH) continue;
      #pragma unroll
      for (int kw = -1; kw <= 1; ++kw) {
        int w2 = w + kw; if (w2 < 0 || w2 >= HH) continue;
        acc += bf2f(y[((size_t)(b*NTOK + h2*HH + w2))*HIDDEN + c]) *
               dww[c*9 + (kh+1)*3 + (kw+1)];
      }
    }
    float ge = 0.5f * acc * (1.0f + erff(acc * 0.70710678118654752f));
    g[(size_t)bn*HIDDEN + c] = f2bf(ge);
  }
}

// ---------------- host launch ----------------
extern "C" void kernel_launch(void* const* d_in, const int* in_sizes, int n_in,
                              void* d_out, int out_size, void* d_ws, size_t ws_size,
                              hipStream_t stream) {
  const float* feat0 = (const float*)d_in[0];
  const float* feat1 = (const float*)d_in[1];
  const float* n1g   = (const float*)d_in[2];
  const float* n1b   = (const float*)d_in[3];
  const float* n2g   = (const float*)d_in[4];
  const float* n2b   = (const float*)d_in[5];
  const float* qw    = (const float*)d_in[6];
  const float* qbb   = (const float*)d_in[7];
  const float* kvw   = (const float*)d_in[8];
  const float* kvbb  = (const float*)d_in[9];
  const float* corw  = (const float*)d_in[10];
  const float* corb  = (const float*)d_in[11];
  const float* motw  = (const float*)d_in[12];
  const float* motb  = (const float*)d_in[13];
  const float* projw = (const float*)d_in[14];
  const float* projb = (const float*)d_in[15];
  const float* fc1w  = (const float*)d_in[16];
  const float* fc1b  = (const float*)d_in[17];
  const float* dww   = (const float*)d_in[18];
  const float* dwb   = (const float*)d_in[19];
  const float* fc2w  = (const float*)d_in[20];
  const float* fc2b  = (const float*)d_in[21];

  float* outp    = (float*)d_out;
  float* motionp = outp + (size_t)BB*CDIM*NTOK;

  char* ws = (char*)d_ws; size_t off = 0;
  auto carve = [&](size_t bytes) -> void* {
    size_t p = (off + 255) & ~(size_t)255; off = p + bytes; return (void*)(ws + p);
  };
  float*  Xf     = (float*) carve((size_t)MROWS*CDIM*4);
  bf16_t* Xb     = (bf16_t*)carve((size_t)MROWS*CDIM*2);
  bf16_t* qwT    = (bf16_t*)carve((size_t)CDIM*CDIM*2);
  bf16_t* kvT    = (bf16_t*)carve((size_t)512*CDIM*2);
  bf16_t* projT  = (bf16_t*)carve((size_t)CDIM*CDIM*2);
  bf16_t* fc1T   = (bf16_t*)carve((size_t)HIDDEN*CDIM*2);
  bf16_t* fc2T   = (bf16_t*)carve((size_t)CDIM*HIDDEN*2);
  bf16_t* motT   = (bf16_t*)carve((size_t)MDIM*MDIM*2);
  bf16_t* Qbuf   = (bf16_t*)carve((size_t)MROWS*CDIM*2);
  bf16_t* KVb    = (bf16_t*)carve((size_t)MROWS*512*2);
  float*  ceb    = (float*) carve((size_t)NTOK*MDIM*4);
  bf16_t* Vt     = (bf16_t*)carve((size_t)BB*NH*48*NTOK*2);
  bf16_t* attnx  = (bf16_t*)carve((size_t)MROWS*CDIM*2);
  bf16_t* crevm  = (bf16_t*)carve((size_t)MROWS*MDIM*2);
  float*  featS  = (float*) carve((size_t)MROWS*CDIM*4);
  bf16_t* ln2b   = (bf16_t*)carve((size_t)MROWS*CDIM*2);
  bf16_t* yb     = (bf16_t*)carve((size_t)MROWS*HIDDEN*2);
  bf16_t* gb     = (bf16_t*)carve((size_t)MROWS*HIDDEN*2);

  // weight prep (bf16, N x K)
  k_transpose<<<(CDIM*CDIM+255)/256,   256, 0, stream>>>(qw,   qwT,  CDIM, CDIM);
  k_transpose<<<(CDIM*512+255)/256,    256, 0, stream>>>(kvw,  kvT,  CDIM, 512);
  k_transpose<<<(CDIM*CDIM+255)/256,   256, 0, stream>>>(projw,projT,CDIM, CDIM);
  k_transpose<<<(CDIM*HIDDEN+255)/256, 256, 0, stream>>>(fc1w, fc1T, CDIM, HIDDEN);
  k_transpose<<<(HIDDEN*CDIM+255)/256, 256, 0, stream>>>(fc2w, fc2T, HIDDEN, CDIM);
  k_transpose<<<(MDIM*MDIM+255)/256,   256, 0, stream>>>(motw, motT, MDIM, MDIM);

  // tokens + LN1
  k_norm_tokens<<<MROWS, 256, 0, stream>>>(feat0, feat1, n1g, n1b, Xf, Xb);

  // Q / KV projections  (grid: x = ceil(Nn/256), y = M/64)
  k_gemm<<<dim3((CDIM+255)/256, MROWS/64), 256, 0, stream>>>(Xb, qwT, qbb, nullptr,
      nullptr, Qbuf, nullptr, MROWS, CDIM, CDIM, NTOK);
  k_gemm<<<dim3((512+255)/256, MROWS/64), 256, 0, stream>>>(Xb, kvT, kvbb, nullptr,
      nullptr, KVb, nullptr, MROWS, 512, CDIM, NTOK);

  // coordinate embedding + V/ce transpose
  k_ce<<<NTOK, 128, 0, stream>>>(corw, corb, ceb);
  k_build_vt<<<BB*NH*48, 256, 0, stream>>>(KVb, ceb, Vt);

  // fused flash attention + motion gather
  k_attn<<<dim3(NTOK/128, NH, BB), 256, 0, stream>>>(Qbuf, KVb, Vt, ceb, attnx, crevm);

  // proj + residual -> feat_s (f32)
  k_gemm<<<dim3((CDIM+255)/256, MROWS/64), 256, 0, stream>>>(attnx, projT, projb, Xf,
      featS, nullptr, nullptr, MROWS, CDIM, CDIM, NTOK);

  // MLP
  k_ln2<<<MROWS, 256, 0, stream>>>(featS, n2g, n2b, ln2b);
  k_gemm<<<dim3((HIDDEN+255)/256, MROWS/64), 256, 0, stream>>>(ln2b, fc1T, fc1b, nullptr,
      nullptr, yb, nullptr, MROWS, HIDDEN, CDIM, NTOK);
  k_dwconv_gelu<<<MROWS, 256, 0, stream>>>(yb, dww, dwb, gb);
  k_gemm<<<dim3((CDIM+255)/256, MROWS/64), 256, 0, stream>>>(gb, fc2T, fc2b, featS,
      nullptr, nullptr, outp, MROWS, CDIM, HIDDEN, NTOK);

  // motion head
  k_gemm<<<dim3((MDIM+255)/256, MROWS/64), 256, 0, stream>>>(crevm, motT, motb, nullptr,
      nullptr, nullptr, motionp, MROWS, MDIM, MDIM, NTOK);
}
